// PhoenixMambaV2_21663815041550
// MI455X (gfx1250) — matrix operand order repaired and checked by reading, not compiled, not run on hardware
//
#include <hip/hip_runtime.h>
#include <hip/hip_bf16.h>

typedef __attribute__((ext_vector_type(16))) _Float16 v16h;
typedef __attribute__((ext_vector_type(8)))  _Float16 h8v;
typedef __attribute__((ext_vector_type(4)))  _Float16 h4v;
typedef __attribute__((ext_vector_type(8)))  float    v8f;
typedef int i4v __attribute__((vector_size(16)));

#if __has_builtin(__builtin_amdgcn_global_load_async_to_lds_b128)
#define USE_ASYNC_LDS 1
#define TO_GLOBAL(p) ((__attribute__((address_space(1))) i4v*)(unsigned long long)(p))
#define TO_LDS(p)    ((__attribute__((address_space(3))) i4v*)(unsigned int)(unsigned long long)(p))
#endif

// ---------------------------------------------------------------------------
// WMMA core: one 16x16x32 MAC from LDS tiles.
//   lAw: per-wave A tile, 16 rows x 64 K, row-major halves.
//   lBt: B tile, 16 cols(N) x 64 K, N-major halves (so K is contiguous).
//   jk : which 32-wide K slice of the 64-wide staged block (0 or 1).
// Fragment layouts per CDNA5 ISA 7.12.2.
// ---------------------------------------------------------------------------
__device__ __forceinline__ void wmma_from_lds(const _Float16* __restrict__ lAw,
                                              const _Float16* __restrict__ lBt,
                                              int jk, v8f& acc) {
    const int lane = threadIdx.x & 31;
    const int hi   = lane >> 4;        // lane group selects K half
    const int mRow = lane & 15;
    const int nCol = lane & 15;
    // A: e<8 -> K = 32jk + 8hi + e ; e>=8 -> K = 32jk + 16 + 8hi + (e-8)
    h8v alo = *(const h8v*)(lAw + mRow * 64 + jk * 32 + hi * 8);
    h8v ahi = *(const h8v*)(lAw + mRow * 64 + jk * 32 + 16 + hi * 8);
    v16h a = __builtin_shufflevector(alo, ahi, 0,1,2,3,4,5,6,7,8,9,10,11,12,13,14,15);
    // B: e<8 -> K = 32jk + 8hi + e ; e>=8 -> K = 32jk + 16 + 8hi + (e-8)
    const h8v* bp = (const h8v*)(lBt + nCol * 64 + jk * 32 + hi * 8);
    v16h b = __builtin_shufflevector(bp[0], bp[2], 0,1,2,3,4,5,6,7,8,9,10,11,12,13,14,15);
    acc = __builtin_amdgcn_wmma_f32_16x16x32_f16(false, a, false, b,
                                                 (short)0, acc, false, false);
}

// Stage one 16(N) x 64(K) f16 B tile from packed [N][Kpad] global into LDS.
// 128 threads, one 16B chunk each. Async global->LDS when available.
__device__ __forceinline__ void stage_b_tile(const _Float16* __restrict__ Bb,
                                             _Float16* __restrict__ lB,
                                             int nBase, int k0, int Kpad) {
    const int i = threadIdx.x;           // 0..127
    const int n = i >> 3, ch = i & 7;
    const _Float16* gsrc = Bb + (long)(nBase + n) * Kpad + k0 + ch * 8;
    _Float16* ldst = lB + n * 64 + ch * 8;
#ifdef USE_ASYNC_LDS
    __builtin_amdgcn_global_load_async_to_lds_b128(TO_GLOBAL(gsrc), TO_LDS(ldst), 0, 0);
    asm volatile("s_wait_asynccnt 0" ::: "memory");
#else
    *(float4*)ldst = *(const float4*)gsrc;
#endif
}

// ---------------------------------------------------------------------------
// Pack an arbitrary strided f32 view into f16 [batch][N][Kpad] (B operand),
// zero-padded in K so GEMM needs no K guards.
// ---------------------------------------------------------------------------
__global__ void pack_b_f16(const float* __restrict__ src, _Float16* __restrict__ dst,
                           int K, int Kpad, int N, long kStride, long nStride,
                           int batchInner, long sOuter, long sInner, int nBatch) {
    const long NK = (long)N * Kpad;
    const long total = NK * nBatch;
    for (long i = (long)blockIdx.x * blockDim.x + threadIdx.x; i < total;
         i += (long)gridDim.x * blockDim.x) {
        long z = i / NK;
        long rem = i - z * NK;
        int n = (int)(rem / Kpad), k = (int)(rem % Kpad);
        long zo = z / batchInner, zi = z % batchInner;
        float v = (k < K) ? src[zo * sOuter + zi * sInner + k * kStride + n * nStride] : 0.f;
        dst[i] = (_Float16)v;
    }
}

// ---------------------------------------------------------------------------
// Generic GEMM: C = act(scale * (A @ B) + bias).  B packed f16 [z][N][Kpad].
// Block = 128 threads = 4 waves; wave w owns rows [bx*64 + w*16, +16).
// Requires: M % 64 == 0, N % 16 == 0, K % 64 == 0 (all uses satisfy this).
// ---------------------------------------------------------------------------
__global__ void gemm_wmma(const float* __restrict__ A, const _Float16* __restrict__ Bp,
                          const float* __restrict__ bias, float* __restrict__ C,
                          int M, int N, int K, int Kpad, int lda, int ldc,
                          int batchInner, long sAo, long sAi, long sCo, long sCi,
                          int act, float scale) {
    __shared__ __align__(16) _Float16 lB[16 * 64];
    __shared__ __align__(16) _Float16 lA[4][16 * 64];
    const int wave = threadIdx.x >> 5;
    const int lane = threadIdx.x & 31;
    const int mBase = blockIdx.x * 64 + wave * 16;
    const int nBase = blockIdx.y * 16;
    const int z = blockIdx.z;
    const int zo = z / batchInner, zi = z % batchInner;
    const float* Ab = A + (long)zo * sAo + (long)zi * sAi;
    const _Float16* Bb = Bp + (long)z * N * Kpad;
    float* Cb = C + (long)zo * sCo + (long)zi * sCi;

    v8f acc = {0.f, 0.f, 0.f, 0.f, 0.f, 0.f, 0.f, 0.f};
    for (int k0 = 0; k0 < K; k0 += 64) {
        stage_b_tile(Bb, lB, nBase, k0, Kpad);
        // A tile: 16 rows x 64 K, f32 -> f16, 16B global loads / 8B LDS stores.
#pragma unroll
        for (int j = 0; j < 8; ++j) {
            int c = lane + j * 32;
            int m = c >> 4, ch = c & 15;
            const float4 v = *(const float4*)(Ab + (long)(mBase + m) * lda + k0 + ch * 4);
            h4v hv = {(_Float16)v.x, (_Float16)v.y, (_Float16)v.z, (_Float16)v.w};
            *(h4v*)(&lA[wave][m * 64 + ch * 4]) = hv;
        }
        if (k0 + 64 < K)
            __builtin_prefetch(Ab + (long)(mBase + (lane & 15)) * lda + k0 + 64, 0, 1);
        __syncthreads();
        wmma_from_lds(lA[wave], lB, 0, acc);
        wmma_from_lds(lA[wave], lB, 1, acc);
        __syncthreads();
    }
    const int n = nBase + (lane & 15);
    const float bv = bias ? bias[n] : 0.f;
#pragma unroll
    for (int r = 0; r < 8; ++r) {
        int m = mBase + r + ((lane >> 4) << 3);
        float v = acc[r] * scale + bv;
        if (act == 1) v = fmaxf(v, 0.f);
        else if (act == 2) v = (v > 20.f) ? v : log1pf(__expf(v));   // softplus
        Cb[(long)m * ldc + n] = v;
    }
}

// ---------------------------------------------------------------------------
// 3x3 SAME conv as implicit GEMM over K = 9*Cin (Cin power of two), NHWC.
// Weights pre-packed f16 [Cout][Kpad]. Branch-free im2col via clamp+mask.
// ---------------------------------------------------------------------------
__global__ void conv3x3_wmma(const float* __restrict__ X, const _Float16* __restrict__ Wp,
                             const float* __restrict__ bias, float* __restrict__ Y,
                             int H, int W, int Cin, int cinShift, int Cout, int relu) {
    __shared__ __align__(16) _Float16 lB[16 * 64];
    __shared__ __align__(16) _Float16 lA[4][16 * 64];
    const int wave = threadIdx.x >> 5;
    const int lane = threadIdx.x & 31;
    const int mBase = blockIdx.x * 64 + wave * 16;   // pixel base within image
    const int nBase = blockIdx.y * 16;               // out-channel base
    const int b = blockIdx.z;
    const float* Xb = X + (long)b * H * W * Cin;
    float* Yb = Y + (long)b * H * W * Cout;
    const int K = 9 * Cin;
    const int Kpad = (K + 63) & ~63;

    v8f acc = {0.f, 0.f, 0.f, 0.f, 0.f, 0.f, 0.f, 0.f};
    for (int k0 = 0; k0 < Kpad; k0 += 64) {
        stage_b_tile(Wp, lB, nBase, k0, Kpad);
        // im2col A tile: branch-free (clamped gather * mask)
        for (int i = lane; i < 1024; i += 32) {
            int m = i >> 6, kk = i & 63;
            int kg = k0 + kk;
            int pix = mBase + m;
            int py = pix / W, px = pix - py * W;
            int tap = kg >> cinShift;           // 0..8 valid, >8 when padded
            int ci = kg & (Cin - 1);
            int oy = py + tap / 3 - 1;
            int ox = px + tap % 3 - 1;
            bool inb = (kg < K) && (oy >= 0) && (oy < H) && (ox >= 0) && (ox < W);
            int oyc = min(max(oy, 0), H - 1);
            int oxc = min(max(ox, 0), W - 1);
            float v = Xb[((long)oyc * W + oxc) * Cin + ci] * (inb ? 1.f : 0.f);
            lA[wave][m * 64 + kk] = (_Float16)v;
        }
        __syncthreads();
        wmma_from_lds(lA[wave], lB, 0, acc);
        wmma_from_lds(lA[wave], lB, 1, acc);
        __syncthreads();
    }
    const int n = nBase + (lane & 15);
    const float bv = bias[n];
#pragma unroll
    for (int r = 0; r < 8; ++r) {
        int pix = mBase + r + ((lane >> 4) << 3);
        float v = acc[r] + bv;
        if (relu) v = fmaxf(v, 0.f);
        Yb[(long)pix * Cout + n] = v;
    }
}

// ---------------------------------------------------------------------------
// LayerNorm over last dim C (one row per block).
// ---------------------------------------------------------------------------
__global__ void layernorm_kernel(const float* __restrict__ x, const float* __restrict__ g,
                                 const float* __restrict__ b, float* __restrict__ out,
                                 int C, float eps) {
    const long row = blockIdx.x;
    const float* xr = x + row * (long)C;
    float* orow = out + row * (long)C;
    __shared__ float red[256];
    float s = 0.f;
    for (int c = threadIdx.x; c < C; c += blockDim.x) s += xr[c];
    red[threadIdx.x] = s; __syncthreads();
    for (int off = 128; off > 0; off >>= 1) {
        if ((int)threadIdx.x < off) red[threadIdx.x] += red[threadIdx.x + off];
        __syncthreads();
    }
    const float mean = red[0] / C; __syncthreads();
    s = 0.f;
    for (int c = threadIdx.x; c < C; c += blockDim.x) { float d = xr[c] - mean; s += d * d; }
    red[threadIdx.x] = s; __syncthreads();
    for (int off = 128; off > 0; off >>= 1) {
        if ((int)threadIdx.x < off) red[threadIdx.x] += red[threadIdx.x + off];
        __syncthreads();
    }
    const float rstd = rsqrtf(red[0] / C + eps); __syncthreads();
    for (int c = threadIdx.x; c < C; c += blockDim.x)
        orow[c] = (xr[c] - mean) * rstd * g[c] + b[c];
}

__global__ void negexp_kernel(const float* __restrict__ in, float* __restrict__ out, int n) {
    int i = blockIdx.x * blockDim.x + threadIdx.x;
    if (i < n) out[i] = -__expf(in[i]);
}

// ---------------------------------------------------------------------------
// S6 selective scan: one thread per (b, d), 16 states kept in registers.
// ---------------------------------------------------------------------------
__global__ void s6_scan_kernel(const float* __restrict__ xln, const float* __restrict__ delta,
                               const float* __restrict__ Bt, const float* __restrict__ Ct,
                               const float* __restrict__ Aneg, const float* __restrict__ Dskip,
                               float* __restrict__ y, int L, int D) {
    const int tid = blockIdx.x * blockDim.x + threadIdx.x;
    const int b = tid / D, d = tid - b * D;
    float A[16], h[16];
#pragma unroll
    for (int n = 0; n < 16; ++n) { A[n] = Aneg[d * 16 + n]; h[n] = 0.f; }
    const float Dd = Dskip[d];
    const float* xr = xln   + (long)b * L * D + d;
    const float* dr = delta + (long)b * L * D + d;
    const float* br = Bt    + (long)b * L * 16;
    const float* cr = Ct    + (long)b * L * 16;
    float* yr = y + (long)b * L * D + d;
    for (int t = 0; t < L; ++t) {
        const float dl = dr[(long)t * D];
        const float xv = xr[(long)t * D];
        const float dx = dl * xv;
        float accum = 0.f;
#pragma unroll
        for (int n = 0; n < 16; ++n) {
            float a = __expf(dl * A[n]);
            h[n] = a * h[n] + dx * br[t * 16 + n];
            accum += h[n] * cr[t * 16 + n];
        }
        yr[(long)t * D] = accum + Dd * xv;
    }
}

__global__ void add_kernel(const float* __restrict__ a, const float* __restrict__ b,
                           float* __restrict__ c, long n) {
    long i = (long)blockIdx.x * blockDim.x + threadIdx.x;
    if (i < n) c[i] = a[i] + b[i];
}

__global__ void maxpool2_kernel(const float* __restrict__ x, float* __restrict__ y,
                                int H, int W, int C, long total) {
    long i = (long)blockIdx.x * blockDim.x + threadIdx.x;
    if (i >= total) return;
    const int Wo = W / 2, Ho = H / 2;
    int c = (int)(i % C); long r = i / C;
    int xo = (int)(r % Wo); r /= Wo;
    int yo = (int)(r % Ho); int b = (int)(r / Ho);
    const float* xb = x + (long)b * H * W * C;
    float v = xb[((long)(2 * yo) * W + 2 * xo) * C + c];
    v = fmaxf(v, xb[((long)(2 * yo) * W + 2 * xo + 1) * C + c]);
    v = fmaxf(v, xb[((long)(2 * yo + 1) * W + 2 * xo) * C + c]);
    v = fmaxf(v, xb[((long)(2 * yo + 1) * W + 2 * xo + 1) * C + c]);
    y[i] = v;
}

__global__ void softmax_rows_kernel(float* __restrict__ x, int C) {
    const long row = blockIdx.x;
    float* xr = x + row * (long)C;
    __shared__ float red[256];
    float m = -1e30f;
    for (int c = threadIdx.x; c < C; c += blockDim.x) m = fmaxf(m, xr[c]);
    red[threadIdx.x] = m; __syncthreads();
    for (int off = 128; off > 0; off >>= 1) {
        if ((int)threadIdx.x < off) red[threadIdx.x] = fmaxf(red[threadIdx.x], red[threadIdx.x + off]);
        __syncthreads();
    }
    m = red[0]; __syncthreads();
    float s = 0.f;
    for (int c = threadIdx.x; c < C; c += blockDim.x) { float e = __expf(xr[c] - m); xr[c] = e; s += e; }
    red[threadIdx.x] = s; __syncthreads();
    for (int off = 128; off > 0; off >>= 1) {
        if ((int)threadIdx.x < off) red[threadIdx.x] += red[threadIdx.x + off];
        __syncthreads();
    }
    const float inv = 1.f / red[0]; __syncthreads();
    for (int c = threadIdx.x; c < C; c += blockDim.x) xr[c] *= inv;
}

__global__ void meanpool_kernel(const float* __restrict__ x, float* __restrict__ out,
                                int T, int C) {
    int i = blockIdx.x * blockDim.x + threadIdx.x;   // B*C threads exactly
    int b = i / C, c = i - b * C;
    float s = 0.f;
    for (int t = 0; t < T; ++t) s += x[((long)b * T + t) * C + c];
    out[i] = s / (float)T;
}

__global__ void cls_softmax_kernel(const float* __restrict__ pooled, const float* __restrict__ w,
                                   const float* __restrict__ bias, float* __restrict__ out) {
    __shared__ float lg[16];
    const int t = threadIdx.x;
    if (t < 16) {
        int b = t >> 2, j = t & 3;
        float s = bias[j];
        for (int c = 0; c < 512; ++c) s += pooled[b * 512 + c] * w[c * 4 + j];
        lg[t] = s;
    }
    __syncthreads();
    if (t < 4) {
        float m = fmaxf(fmaxf(lg[t * 4], lg[t * 4 + 1]), fmaxf(lg[t * 4 + 2], lg[t * 4 + 3]));
        float e0 = __expf(lg[t * 4 + 0] - m), e1 = __expf(lg[t * 4 + 1] - m);
        float e2 = __expf(lg[t * 4 + 2] - m), e3 = __expf(lg[t * 4 + 3] - m);
        float inv = 1.f / (e0 + e1 + e2 + e3);
        out[t * 4 + 0] = e0 * inv; out[t * 4 + 1] = e1 * inv;
        out[t * 4 + 2] = e2 * inv; out[t * 4 + 3] = e3 * inv;
    }
}

// ---------------------------------------------------------------------------
// Host orchestration
// ---------------------------------------------------------------------------
static const int STAGE_DIMS_H[4] = {64, 128, 256, 512};
static const int CONV_IN_H[4]    = {64, 64, 128, 256};
static const int CONV_SHIFT_H[4] = {6, 6, 7, 8};

extern "C" void kernel_launch(void* const* d_in, const int* in_sizes, int n_in,
                              void* d_out, int out_size, void* d_ws, size_t ws_size,
                              hipStream_t stream) {
    (void)in_sizes; (void)n_in; (void)out_size; (void)ws_size;
    const int B = 4;

    // ---- workspace layout (floats) ----
    float* f     = (float*)d_ws;
    float* act0  = f;                    // 4,194,304
    float* act1  = act0 + 4194304;       // 4,194,304
    float* lnb   = act1 + 4194304;       // 4,194,304
    float* dlt   = lnb  + 4194304;       // 4,194,304
    float* ybuf  = dlt  + 4194304;       // 4,194,304
    float* btb   = ybuf + 4194304;       // 1,048,576
    float* ctb   = btb  + 1048576;       // 1,048,576
    float* aneg  = ctb  + 1048576;       //     8,192
    float* qb    = aneg + 8192;          //   524,288
    float* kb    = qb   + 524288;        //   524,288
    float* vb    = kb   + 524288;        //   524,288
    float* sb    = vb   + 524288;        // 1,048,576
    float* ctxb  = sb   + 1048576;       //   524,288
    float* poolb = ctxb + 524288;        //     2,048
    _Float16* hb = (_Float16*)(poolb + 2048);   // packed f16 B operands (<= ~1.2M halves)

    auto kpad64 = [](int K) { return (K + 63) & ~63; };
    auto packW = [&](const float* src, int K, int N) {   // plain weight (K rows, N cols)
        pack_b_f16<<<dim3(2048), 256, 0, stream>>>(src, hb, K, kpad64(K), N,
                                                   (long)N, 1L, 1, 0L, 0L, 1);
    };
    auto gemm = [&](const float* A, const float* bias, float* C, int M, int N, int K,
                    int lda, int ldc, int act, float scale) {
        gemm_wmma<<<dim3(M / 64, N / 16, 1), 128, 0, stream>>>(
            A, hb, bias, C, M, N, K, kpad64(K), lda, ldc, 1, 0L, 0L, 0L, 0L, act, scale);
    };

    const float* input = (const float*)d_in[0];

    // ---- SliceAggregator: conv 16 -> 64 @128x128, ReLU ----
    packW((const float*)d_in[1], 9 * 16, 64);
    conv3x3_wmma<<<dim3(128 * 128 / 64, 64 / 16, B), 128, 0, stream>>>(
        input, hb, (const float*)d_in[2], act0, 128, 128, 16, 4, 64, 1);

    float* cur = act0;
    float* other = act1;

    // ---- 4 stages ----
    for (int i = 0; i < 4; ++i) {
        const int S = 128 >> i;
        const int Cin = CONV_IN_H[i];
        const int D = STAGE_DIMS_H[i];
        const int base = 3 + i * 10;
        const float* cw   = (const float*)d_in[base + 0];
        const float* cb   = (const float*)d_in[base + 1];
        const float* lng  = (const float*)d_in[base + 2];
        const float* lnbt = (const float*)d_in[base + 3];
        const float* Wd   = (const float*)d_in[base + 4];
        const float* bd   = (const float*)d_in[base + 5];
        const float* WB   = (const float*)d_in[base + 6];
        const float* WC   = (const float*)d_in[base + 7];
        const float* Alog = (const float*)d_in[base + 8];
        const float* Dsk  = (const float*)d_in[base + 9];
        const long Lh = (long)S * S;
        const int M = (int)(B * Lh);

        // conv (no activation) -> `other` holds residual
        packW(cw, 9 * Cin, D);
        conv3x3_wmma<<<dim3((int)(Lh / 64), D / 16, B), 128, 0, stream>>>(
            cur, hb, cb, other, S, S, Cin, CONV_SHIFT_H[i], D, 0);

        // layernorm
        layernorm_kernel<<<dim3(M), 256, 0, stream>>>(other, lng, lnbt, lnb, D, 1e-5f);

        // delta = softplus(ln @ Wd + bd)
        packW(Wd, D, D);
        gemm(lnb, bd, dlt, M, D, D, D, D, 2, 1.f);
        // Bt = ln @ WB ; Ct = ln @ WC
        packW(WB, D, 16);
        gemm(lnb, nullptr, btb, M, 16, D, D, 16, 0, 1.f);
        packW(WC, D, 16);
        gemm(lnb, nullptr, ctb, M, 16, D, D, 16, 0, 1.f);

        // A = -exp(A_log)
        negexp_kernel<<<dim3((D * 16 + 255) / 256), 256, 0, stream>>>(Alog, aneg, D * 16);

        // selective scan
        s6_scan_kernel<<<dim3(B * D / 64), 64, 0, stream>>>(
            lnb, dlt, btb, ctb, aneg, Dsk, ybuf, (int)Lh, D);

        // residual add: cur = conv_out + s6_out
        {
            long n = (long)M * D;
            add_kernel<<<dim3((int)((n + 255) / 256)), 256, 0, stream>>>(other, ybuf, cur, n);
        }

        if (i < 3) {
            long total = (long)B * (S / 2) * (S / 2) * D;
            maxpool2_kernel<<<dim3((int)((total + 255) / 256)), 256, 0, stream>>>(
                cur, other, S, S, D, total);
            float* t = cur; cur = other; other = t;
        }
    }

    // ---- attention on (B, 256, 512); cur holds xf ----
    const int T = 256, DM = 512, HD = 128, NH = 4;
    const int Mq = B * T;
    layernorm_kernel<<<dim3(Mq), 256, 0, stream>>>(
        cur, (const float*)d_in[43], (const float*)d_in[44], lnb, DM, 1e-3f);

    packW((const float*)d_in[45], DM, DM);
    gemm(lnb, (const float*)d_in[46], qb, Mq, DM, DM, DM, DM, 0, 1.f);
    packW((const float*)d_in[47], DM, DM);
    gemm(lnb, (const float*)d_in[48], kb, Mq, DM, DM, DM, DM, 0, 1.f);
    packW((const float*)d_in[49], DM, DM);
    gemm(lnb, (const float*)d_in[50], vb, Mq, DM, DM, DM, DM, 0, 1.f);

    // scores: S[bh] = Q_h (256x128) @ K_h^T (128x256) / sqrt(128)
    pack_b_f16<<<dim3(2048), 256, 0, stream>>>(
        kb, hb, HD, HD, T, 1L, (long)DM, NH, (long)T * DM, (long)HD, B * NH);
    gemm_wmma<<<dim3(T / 64, T / 16, B * NH), 128, 0, stream>>>(
        qb, hb, nullptr, sb, T, T, HD, HD, DM, T,
        NH, (long)T * DM, (long)HD, (long)NH * T * T, (long)T * T,
        0, 0.08838834764831845f);

    softmax_rows_kernel<<<dim3(B * NH * T), 256, 0, stream>>>(sb, T);

    // context: O[bh] = P (256x256) @ V_h (256x128), scattered back to (B,T,H,hd)
    pack_b_f16<<<dim3(2048), 256, 0, stream>>>(
        vb, hb, T, T, HD, (long)DM, 1L, NH, (long)T * DM, (long)HD, B * NH);
    gemm_wmma<<<dim3(T / 64, HD / 16, B * NH), 128, 0, stream>>>(
        sb, hb, nullptr, ctxb, T, HD, T, T, T, DM,
        NH, (long)NH * T * T, (long)T * T, (long)T * DM, (long)HD,
        0, 1.f);

    // output projection + residual
    packW((const float*)d_in[51], DM, DM);
    gemm(ctxb, (const float*)d_in[52], ybuf, Mq, DM, DM, DM, DM, 0, 1.f);
    {
        long n = (long)Mq * DM;
        add_kernel<<<dim3((int)((n + 255) / 256)), 256, 0, stream>>>(cur, ybuf, other, n);
    }

    // mean pool over T, then classifier + softmax -> d_out (4x4 f32)
    meanpool_kernel<<<dim3(B * DM / 256), 256, 0, stream>>>(other, poolb, T, DM);
    cls_softmax_kernel<<<dim3(1), 32, 0, stream>>>(
        poolb, (const float*)d_in[53], (const float*)d_in[54], (float*)d_out);
}
